// DiffusionTextModel_3478923509920
// MI455X (gfx1250) — matrix operand order repaired and checked by reading, not compiled, not run on hardware
//
#include <hip/hip_runtime.h>
#include <hip/hip_bf16.h>
#include <math.h>

// ---------------------------------------------------------------------------
// DiffusionTextModel forward for MI455X (gfx1250, wave32, WMMA).
// f32 in HBM, bf16 tiles in LDS, v_wmma_f32_16x16x32_bf16 compute.
// 128x128 block tile, 256 threads (8 waves), wave tile 64x32 (4x2 WMMA accs).
// Software-pipelined: next tile's global loads overlap current tile's WMMAs.
// ---------------------------------------------------------------------------

typedef __attribute__((ext_vector_type(16))) __bf16 v16bf;
typedef __attribute__((ext_vector_type(8)))  __bf16 v8bf;
typedef __attribute__((ext_vector_type(4)))  __bf16 v4bf;
typedef __attribute__((ext_vector_type(8)))  float  v8f;

#define NT     4096      // B*S tokens
#define DLAT   1024
#define NEXP   16
#define VOCAB  32000

#define ACT_NONE 0
#define ACT_RELU 1
#define ACT_SILU 2

#define LDS_STRIDE 48    // bf16 elements per LDS row (96B, 16B aligned)

// C[m, n] = act( sum_k A[m, k] * W[k, n] + bias[n] )
// A: up to 3 row-major [M x 1024] segments concatenated along K.
// W: row-major [Ktot x Ntot].  C row stride = ldc.
__global__ __launch_bounds__(256)
void gemm_wmma_bf16(const float* __restrict__ A0,
                    const float* __restrict__ A1,
                    const float* __restrict__ A2,
                    const float* __restrict__ W,
                    const float* __restrict__ bias,
                    float* __restrict__ C,
                    int Ntot, int Ktot, int ldc, int act)
{
    __shared__ __bf16 Asm[128][LDS_STRIDE];   // [m][k]
    __shared__ __bf16 Bsm[128][LDS_STRIDE];   // transposed: [n][k]

    const int tid  = threadIdx.x;             // 0..255
    const int lane = tid & 31;
    const int wave = tid >> 5;                // 0..7
    const int half = lane >> 4;               // 0/1
    const int lr   = lane & 15;
    const int row_off = (wave & 1) * 64;      // 2 wave-rows of 64
    const int col_off = (wave >> 1) * 32;     // 4 wave-cols of 32
    const int bm = blockIdx.x * 128;
    const int bn = blockIdx.y * 128;

    // Staging coordinates
    const int ar0 = tid >> 3;                 // A: base row 0..31 (p adds 32)
    const int ac  = (tid & 7) * 4;            // A: col group (float4)
    const int bnl = tid & 127;                // B: local n
    const int bkh = (tid >> 7) * 16;          // B: k half (0 or 16)
    const int gn  = bn + bnl;
    const bool nok = gn < Ntot;

    v8f acc[4][2] = {};
    float4 aReg[4];
    float  bReg[16];

    // ---- global-load phase (into registers) ----
    auto load_tile = [&](int kt) {
        const float* sp = (kt < 1024) ? A0 : ((kt < 2048) ? A1 : A2);
        const int kb = kt & 1023;
        const float* ab = sp + (size_t)(bm + ar0) * 1024 + kb + ac;
        #pragma unroll
        for (int p = 0; p < 4; ++p)
            aReg[p] = *(const float4*)(ab + (size_t)p * 32 * 1024);
        const float* wb = W + (size_t)(kt + bkh) * Ntot + gn;
        #pragma unroll
        for (int j = 0; j < 16; ++j)
            bReg[j] = nok ? wb[(size_t)j * Ntot] : 0.0f;
    };

    // ---- convert + LDS-store phase ----
    auto store_tile = [&]() {
        #pragma unroll
        for (int p = 0; p < 4; ++p) {
            v4bf v;
            v[0] = (__bf16)aReg[p].x; v[1] = (__bf16)aReg[p].y;
            v[2] = (__bf16)aReg[p].z; v[3] = (__bf16)aReg[p].w;
            *(v4bf*)&Asm[ar0 + p * 32][ac] = v;
        }
        v8bf b0, b1;
        #pragma unroll
        for (int j = 0; j < 8; ++j) {
            b0[j] = (__bf16)bReg[j];
            b1[j] = (__bf16)bReg[j + 8];
        }
        *(v8bf*)&Bsm[bnl][bkh]     = b0;
        *(v8bf*)&Bsm[bnl][bkh + 8] = b1;
    };

    load_tile(0);
    for (int kt = 0; kt < Ktot; kt += 32) {
        store_tile();
        __syncthreads();
        if (kt + 32 < Ktot) load_tile(kt + 32);   // overlap with WMMA below

        // Fragments per ISA 7.12.2 16-bit layouts (wave32)
        v16bf af[4], bfr[2];
        #pragma unroll
        for (int mi = 0; mi < 4; ++mi) {
            int m = row_off + mi * 16 + lr;       // A: M = lane%16
            v8bf lo = *(const v8bf*)&Asm[m][8 * half];       // K = e + 8*half
            v8bf hi = *(const v8bf*)&Asm[m][16 + 8 * half];  // K = 16+e+8*half
            af[mi] = __builtin_shufflevector(lo, hi,
                0,1,2,3,4,5,6,7,8,9,10,11,12,13,14,15);
        }
        #pragma unroll
        for (int ni = 0; ni < 2; ++ni) {
            int n = col_off + ni * 16 + lr;       // B: N = lane%16
            v8bf lo = *(const v8bf*)&Bsm[n][8 * half];       // K = e + 8*half
            v8bf hi = *(const v8bf*)&Bsm[n][16 + 8 * half];
            bfr[ni] = __builtin_shufflevector(lo, hi,
                0,1,2,3,4,5,6,7,8,9,10,11,12,13,14,15);
        }
        #pragma unroll
        for (int mi = 0; mi < 4; ++mi)
            #pragma unroll
            for (int ni = 0; ni < 2; ++ni)
                acc[mi][ni] = __builtin_amdgcn_wmma_f32_16x16x32_bf16(
                    false, af[mi], false, bfr[ni],
                    (short)0, acc[mi][ni], false, false);
        __syncthreads();
    }

    // Epilogue: C layout  M = r + 8*half, N = lane%16
    #pragma unroll
    for (int mi = 0; mi < 4; ++mi) {
        #pragma unroll
        for (int ni = 0; ni < 2; ++ni) {
            int n = bn + col_off + ni * 16 + lr;
            if (n < Ntot) {
                float bv = bias ? bias[n] : 0.0f;
                #pragma unroll
                for (int r = 0; r < 8; ++r) {
                    int m = bm + row_off + mi * 16 + 8 * half + r;
                    float v = acc[mi][ni][r] + bv;
                    if (act == ACT_RELU)      v = v > 0.0f ? v : 0.0f;
                    else if (act == ACT_SILU) v = v / (1.0f + __expf(-v));
                    C[(long)m * ldc + n] = v;
                }
            }
        }
    }
}

// In-place row softmax over 16 elements, one thread per row.
__global__ void softmax16_kernel(float* __restrict__ p, int rows)
{
    int r = blockIdx.x * blockDim.x + threadIdx.x;
    if (r >= rows) return;
    float* x = p + (long)r * NEXP;
    float mx = x[0];
    #pragma unroll
    for (int i = 1; i < NEXP; ++i) mx = fmaxf(mx, x[i]);
    float s = 0.0f;
    float e[NEXP];
    #pragma unroll
    for (int i = 0; i < NEXP; ++i) { e[i] = __expf(x[i] - mx); s += e[i]; }
    float inv = 1.0f / s;
    #pragma unroll
    for (int i = 0; i < NEXP; ++i) x[i] = e[i] * inv;
}

// qk[r, d] = sum_e expert_outputs[r, e, d] * mask[r, e]
__global__ void qk_combine_kernel(const float* __restrict__ expo,
                                  const float* __restrict__ mask,
                                  float* __restrict__ out)
{
    long idx = (long)blockIdx.x * blockDim.x + threadIdx.x;
    if (idx >= (long)NT * DLAT) return;
    int r = (int)(idx >> 10);
    int d = (int)(idx & 1023);
    const float* eb = expo + ((long)r * NEXP) * DLAT + d;
    const float* mb = mask + (long)r * NEXP;
    float s = 0.0f;
    #pragma unroll
    for (int e = 0; e < NEXP; ++e) s += eb[(long)e * DLAT] * mb[e];
    out[idx] = s;
}

extern "C" void kernel_launch(void* const* d_in, const int* in_sizes, int n_in,
                              void* d_out, int out_size, void* d_ws, size_t ws_size,
                              hipStream_t stream)
{
    (void)in_sizes; (void)n_in; (void)out_size; (void)ws_size;

    const float* L      = (const float*)d_in[0];
    const float* mL     = (const float*)d_in[1];
    const float* prompt = (const float*)d_in[2];
    const float* noise  = (const float*)d_in[3];
    const float* qa_W1 = (const float*)d_in[4];  const float* qa_b1 = (const float*)d_in[5];
    const float* qa_W2 = (const float*)d_in[6];  const float* qa_b2 = (const float*)d_in[7];
    const float* qb_W1 = (const float*)d_in[8];  const float* qb_b1 = (const float*)d_in[9];
    const float* qb_W2 = (const float*)d_in[10]; const float* qb_b2 = (const float*)d_in[11];
    const float* qc_W1 = (const float*)d_in[12]; const float* qc_b1 = (const float*)d_in[13];
    const float* qc_W2 = (const float*)d_in[14]; const float* qc_b2 = (const float*)d_in[15];
    const float* ek_W1 = (const float*)d_in[16]; const float* ek_b1 = (const float*)d_in[17];
    const float* ek_W2 = (const float*)d_in[18]; const float* ek_b2 = (const float*)d_in[19];
    const float* md_W1 = (const float*)d_in[20]; const float* md_b1 = (const float*)d_in[21];
    const float* md_W2 = (const float*)d_in[22]; const float* md_b2 = (const float*)d_in[23];
    const float* td_W  = (const float*)d_in[24]; const float* td_b  = (const float*)d_in[25];

    // Output layout (flat concat, f32): L2, mL_pred, mask, text_logits, expert_outputs
    float* out  = (float*)d_out;
    float* L2   = out;
    float* mLp  = out + (size_t)NT * DLAT;
    float* mask = out + (size_t)2 * NT * DLAT;
    float* text = mask + (size_t)NT * NEXP;
    float* expo = text + (size_t)NT * VOCAB;

    // Workspace: 4 x [NT x D] f32 slices = 64 MB
    float* ws = (float*)d_ws;
    float* h  = ws;                          // denoiser / expert hidden
    float* L1 = h  + (size_t)NT * DLAT;
    float* sL = L1 + (size_t)NT * DLAT;
    float* t  = sL + (size_t)NT * DLAT;      // mask hidden; later reused as qk_output

    dim3 blk(256);
    dim3 gD(NT / 128, DLAT / 128);           // N = 1024
    dim3 gE(NT / 128, 1);                    // N = 16
    dim3 gV(NT / 128, VOCAB / 128);          // N = 32000

    // L1 = qa([L | prompt | mL])
    gemm_wmma_bf16<<<gD, blk, 0, stream>>>(L, prompt, mL, qa_W1, qa_b1, h,
                                           DLAT, 3072, DLAT, ACT_RELU);
    gemm_wmma_bf16<<<gD, blk, 0, stream>>>(h, nullptr, nullptr, qa_W2, qa_b2, L1,
                                           DLAT, 1024, DLAT, ACT_NONE);
    // sL = qb([noise | prompt | L1])
    gemm_wmma_bf16<<<gD, blk, 0, stream>>>(noise, prompt, L1, qb_W1, qb_b1, h,
                                           DLAT, 3072, DLAT, ACT_RELU);
    gemm_wmma_bf16<<<gD, blk, 0, stream>>>(h, nullptr, nullptr, qb_W2, qb_b2, sL,
                                           DLAT, 1024, DLAT, ACT_NONE);
    // mask = softmax(relu(sL @ md_W1) @ md_W2)
    gemm_wmma_bf16<<<gD, blk, 0, stream>>>(sL, nullptr, nullptr, md_W1, md_b1, t,
                                           DLAT, 1024, DLAT, ACT_RELU);
    gemm_wmma_bf16<<<gE, blk, 0, stream>>>(t, nullptr, nullptr, md_W2, md_b2, mask,
                                           NEXP, 1024, NEXP, ACT_NONE);
    softmax16_kernel<<<(NT + 255) / 256, 256, 0, stream>>>(mask, NT);

    // Dense soft-MoE: all 16 experts, writing straight into expert_outputs
    for (int e = 0; e < NEXP; ++e) {
        const float* w1 = ek_W1 + (size_t)e * DLAT * DLAT;
        const float* b1 = ek_b1 + (size_t)e * DLAT;
        const float* w2 = ek_W2 + (size_t)e * DLAT * DLAT;
        const float* b2 = ek_b2 + (size_t)e * DLAT;
        gemm_wmma_bf16<<<gD, blk, 0, stream>>>(L1, nullptr, nullptr, w1, b1, h,
                                               DLAT, 1024, DLAT, ACT_SILU);
        gemm_wmma_bf16<<<gD, blk, 0, stream>>>(h, nullptr, nullptr, w2, b2,
                                               expo + (size_t)e * DLAT,
                                               DLAT, 1024, NEXP * DLAT, ACT_NONE);
    }

    // qk_output = sum_e expert_outputs[:, e, :] * mask[:, e]   (reuse t)
    qk_combine_kernel<<<(int)(((long)NT * DLAT + 255) / 256), 256, 0, stream>>>(expo, mask, t);

    // L2 = qa([qk | prompt | mL])
    gemm_wmma_bf16<<<gD, blk, 0, stream>>>(t, prompt, mL, qa_W1, qa_b1, h,
                                           DLAT, 3072, DLAT, ACT_RELU);
    gemm_wmma_bf16<<<gD, blk, 0, stream>>>(h, nullptr, nullptr, qa_W2, qa_b2, L2,
                                           DLAT, 1024, DLAT, ACT_NONE);
    // mL_pred = qc([mL | prompt | L2])
    gemm_wmma_bf16<<<gD, blk, 0, stream>>>(mL, prompt, L2, qc_W1, qc_b1, h,
                                           DLAT, 3072, DLAT, ACT_RELU);
    gemm_wmma_bf16<<<gD, blk, 0, stream>>>(h, nullptr, nullptr, qc_W2, qc_b2, mLp,
                                           DLAT, 1024, DLAT, ACT_NONE);
    // text_logits = L2 @ td_W + td_b
    gemm_wmma_bf16<<<gV, blk, 0, stream>>>(L2, nullptr, nullptr, td_W, td_b, text,
                                           VOCAB, 1024, VOCAB, ACT_NONE);
}
